// RNN_49297634623575
// MI455X (gfx1250) — matrix operand order repaired and checked by reading, not compiled, not run on hardware
//
#include <hip/hip_runtime.h>
#include <math.h>

typedef __attribute__((ext_vector_type(16))) _Float16 v16h;
typedef __attribute__((ext_vector_type(8)))  _Float16 v8h;
typedef __attribute__((ext_vector_type(2)))  __fp16   fp16x2;  // cvt_pkrtz result type
typedef __attribute__((ext_vector_type(8)))  float    v8f;
typedef __attribute__((ext_vector_type(4)))  float    v4f;

#define RNN_B 65536
#define RNN_T 28
#define RNN_I 28
#define RNN_H 64
#define RNN_C 10
#define HSTR  72   // f16 row stride: 144B = 9*16B, keeps b128 LDS loads aligned

union V16 { v16h v; fp16x2 p[8]; v8h h[2]; };

// Branch-free tanh: prefer the CDNA5 V_TANH_F32 transcendental if declared.
__device__ __forceinline__ float fast_tanh(float x) {
#if __has_builtin(__builtin_amdgcn_tanhf)
  return __builtin_amdgcn_tanhf(x);
#elif __has_builtin(__builtin_amdgcn_tanh_f32)
  return __builtin_amdgcn_tanh_f32(x);
#else
  // tanh(x) = sign(x) * (1 - 2/(1 + exp(2|x|)));  exp via v_exp_f32 (2^y)
  const float ax = __builtin_fabsf(x);
  const float e  = __builtin_amdgcn_exp2f(ax * -2.8853900817779268f); // e^{-2|x|}
  const float t  = (1.0f - e) * __builtin_amdgcn_rcpf(1.0f + e);
  return __builtin_copysignf(t, x);
#endif
}

// A-fragment (16x32 f16) from 4x f32 quads: lane = row M, kc = 8*(lane/16)
// halfs 0..7 <-> K=kc+0..7, halfs 8..15 <-> K=16+kc+0..7
__device__ __forceinline__ v16h pack_afrag(v4f a, v4f b, v4f c, v4f d) {
  V16 u;
  u.p[0] = __builtin_amdgcn_cvt_pkrtz(a[0], a[1]);
  u.p[1] = __builtin_amdgcn_cvt_pkrtz(a[2], a[3]);
  u.p[2] = __builtin_amdgcn_cvt_pkrtz(b[0], b[1]);
  u.p[3] = __builtin_amdgcn_cvt_pkrtz(b[2], b[3]);
  u.p[4] = __builtin_amdgcn_cvt_pkrtz(c[0], c[1]);
  u.p[5] = __builtin_amdgcn_cvt_pkrtz(c[2], c[3]);
  u.p[6] = __builtin_amdgcn_cvt_pkrtz(d[0], d[1]);
  u.p[7] = __builtin_amdgcn_cvt_pkrtz(d[2], d[3]);
  return u.v;
}

__global__ __launch_bounds__(256) void rnn_fused_kernel(
    const float* __restrict__ x,     // [B, T*I]
    const float* __restrict__ W_ih,  // [H, I]
    const float* __restrict__ W_hh,  // [H, H]
    const float* __restrict__ b_ih,  // [H]
    const float* __restrict__ b_hh,  // [H]
    const float* __restrict__ fc_w,  // [C, H]
    const float* __restrict__ fc_b,  // [C]
    float* __restrict__ out)         // [B, C]
{
  __shared__ _Float16 hbuf[8][16][HSTR];  // per-wave 16x64 f16 h tile (padded)

  const int tid  = threadIdx.x;
  const int wave = tid >> 5;
  const int lane = tid & 31;
  const int nlo  = lane & 15;  // A-row within tile / D-column index
  const int lg   = lane >> 4;  // lane group
  const int kc   = lg * 8;     // A-fragment K sub-offset

  const int tile  = blockIdx.x * 8 + wave;
  const int bbase = tile * 16;

  // ---- Preload B fragments = W^T tiles (f16). B half j <-> K = kb+8*lg+j+(j&8),
  // column N = n0 + lane%16, B(k,n) = W[n][k]. Bias folded into K=28 of W_ih.
  v16h wih[4];
  v16h whh[2][4];
#pragma unroll
  for (int nt = 0; nt < 4; ++nt) {
    const int row = nt * 16 + nlo;
    const float brow = b_ih[row] + b_hh[row];
    {
      v16h r;
#pragma unroll
      for (int j = 0; j < 16; ++j) {
        const int k = 8 * lg + j + (j & 8);
        float v = 0.0f;
        if (k < RNN_I) v = W_ih[row * RNN_I + k];
        else if (k == RNN_I) v = brow;          // bias lane: pairs with xA==1.0
        r[j] = (_Float16)v;
      }
      wih[nt] = r;
    }
#pragma unroll
    for (int kt = 0; kt < 2; ++kt) {
      v16h r;
#pragma unroll
      for (int j = 0; j < 16; ++j)
        r[j] = (_Float16)W_hh[row * RNN_H + kt * 32 + 8 * lg + j + (j & 8)];
      whh[kt][nt] = r;
    }
  }

  _Float16* myh = &hbuf[wave][0][0];
  const float* xrow = x + (size_t)(bbase + nlo) * (RNN_T * RNN_I);
  const v4f one0  = {1.0f, 0.0f, 0.0f, 0.0f};  // K=28 -> 1.0 (bias), K=29..31 -> 0
  const v8f zeroC = {};

  v16h hA0 = {};  // h A-fragment, K = 0..31 (h0 = 0)
  v16h hA1 = {};  // h A-fragment, K = 32..63

  for (int t = 0; t < RNN_T; ++t) {
    // x A-fragment (K = 28 real, K=28 carries the bias 1.0, 29..31 zero)
    const float* xp = xrow + t * RNN_I + kc;
    const float* p3 = lg ? xrow : (xp + 20);   // keep address in-bounds for lg=1
    v4f c0 = *(const v4f*)(xp);
    v4f c1 = *(const v4f*)(xp + 4);
    v4f c2 = *(const v4f*)(xp + 16);
    v4f c3 = *(const v4f*)(p3);
    if (lg) c3 = one0;
    v16h xA = pack_afrag(c0, c1, c2, c3);

    v8f acc[4];
#pragma unroll
    for (int nt = 0; nt < 4; ++nt) {
      acc[nt] = __builtin_amdgcn_wmma_f32_16x16x32_f16(
          false, xA, false, wih[nt], (short)0, zeroC, false, false);
      acc[nt] = __builtin_amdgcn_wmma_f32_16x16x32_f16(
          false, hA0, false, whh[0][nt], (short)0, acc[nt], false, false);
      acc[nt] = __builtin_amdgcn_wmma_f32_16x16x32_f16(
          false, hA1, false, whh[1][nt], (short)0, acc[nt], false, false);
    }

    // h = tanh(acc) -> f16 -> LDS row-major (transpose D-layout -> A-layout).
    // D layout: acc[nt][r] is element (M = r + 8*lg, N = nt*16 + lane%16).
#pragma unroll
    for (int nt = 0; nt < 4; ++nt)
#pragma unroll
      for (int r = 0; r < 8; ++r)
        myh[(r + 8 * lg) * HSTR + nt * 16 + nlo] =
            (_Float16)fast_tanh(acc[nt][r]);

    asm volatile("s_wait_dscnt 0" ::: "memory");

    // Reload h A-fragments as raw f16 (no conversions): 4x ds_load_b128
    const _Float16* hr = myh + nlo * HSTR;
    V16 u0, u1;
    u0.h[0] = *(const v8h*)(hr + kc);
    u0.h[1] = *(const v8h*)(hr + 16 + kc);
    u1.h[0] = *(const v8h*)(hr + 32 + kc);
    u1.h[1] = *(const v8h*)(hr + 48 + kc);
    hA0 = u0.v;
    hA1 = u1.v;

    asm volatile("s_wait_dscnt 0" ::: "memory");
  }

  // ---- FC (64 -> 10) + softmax epilogue: lanes 0..15, one batch row each ----
  if (lane < 16) {
    const _Float16* hr = myh + lane * HSTR;  // final h (f16) for this row
    float logits[RNN_C];
#pragma unroll
    for (int c = 0; c < RNN_C; ++c) {
      float s = fc_b[c];
#pragma unroll
      for (int k = 0; k < RNN_H; ++k) s += (float)hr[k] * fc_w[c * RNN_H + k];
      logits[c] = s;
    }
    float mx = logits[0];
#pragma unroll
    for (int c = 1; c < RNN_C; ++c) mx = fmaxf(mx, logits[c]);
    float sum = 0.0f;
#pragma unroll
    for (int c = 0; c < RNN_C; ++c) {
      logits[c] = __builtin_amdgcn_exp2f((logits[c] - mx) * 1.4426950408889634f);
      sum += logits[c];
    }
    const float inv = __builtin_amdgcn_rcpf(sum);
    float* orow = out + (size_t)(bbase + lane) * RNN_C;
#pragma unroll
    for (int c = 0; c < RNN_C; ++c) orow[c] = logits[c] * inv;
  }
}

extern "C" void kernel_launch(void* const* d_in, const int* in_sizes, int n_in,
                              void* d_out, int out_size, void* d_ws, size_t ws_size,
                              hipStream_t stream) {
  const float* x    = (const float*)d_in[0];
  const float* W_ih = (const float*)d_in[1];
  const float* W_hh = (const float*)d_in[2];
  const float* b_ih = (const float*)d_in[3];
  const float* b_hh = (const float*)d_in[4];
  const float* fc_w = (const float*)d_in[5];
  const float* fc_b = (const float*)d_in[6];
  float* out = (float*)d_out;

  // 4096 waves (16 batch rows each), 8 waves per block -> 512 blocks
  dim3 grid(RNN_B / 16 / 8), block(256);
  rnn_fused_kernel<<<grid, block, 0, stream>>>(x, W_ih, W_hh, b_ih, b_hh,
                                               fc_w, fc_b, out);
}
